// SharedAttentionProcessor_guni_28209345200829
// MI455X (gfx1250) — matrix operand order repaired and checked by reading, not compiled, not run on hardware
//
#include <hip/hip_runtime.h>

typedef _Float16 __attribute__((ext_vector_type(16))) v16h;
typedef _Float16 __attribute__((ext_vector_type(8)))  v8h;
typedef float    __attribute__((ext_vector_type(8)))  v8f;

union V16H { v16h v; v8h h[2]; };
union BP16 { v16h v; unsigned u[8]; };

#define WMMA_F16(A, B, C) \
    __builtin_amdgcn_wmma_f32_16x16x32_f16(false, (A), false, (B), (short)0, (C), false, false)

// B=4, N=2048, C=640, HEADS=10, HEAD_DIM=64
static const int NB = 4, NN = 2048, NC = 640, NH = 10, HD = 64;

__device__ inline unsigned pkf16(float a, float b) {
    union { _Float16 f[2]; unsigned u; } t;
    t.f[0] = (_Float16)a; t.f[1] = (_Float16)b;
    return t.u;
}

// ---------------------------------------------------------------- f32 -> f16
__global__ void k_tof16(const float* __restrict__ s, _Float16* __restrict__ d, int n) {
    int i = blockIdx.x * blockDim.x + threadIdx.x;
    if (i < n) d[i] = (_Float16)s[i];
}

// ---------------------------------------------------------------- GEMM: C[M,640] = A[M,640](f16) @ W^T, W row-major [640,640](f16)
// one wave computes a 32(M) x 64(N) tile; FUSE adds bias + residual (f32)
template <bool FUSE>
__global__ void k_gemm(const _Float16* __restrict__ A, const _Float16* __restrict__ W,
                       float* __restrict__ C, int M,
                       const float* __restrict__ bias, const float* __restrict__ resid) {
    const int K = NC;
    int wave = (blockIdx.x * blockDim.x + threadIdx.x) >> 5;
    int lane = threadIdx.x & 31;
    int ntiles = NC / 64;                 // 10
    int mtiles = M / 32;
    if (wave >= mtiles * ntiles) return;  // wave-uniform guard
    int mt = wave / ntiles, nt = wave % ntiles;
    int mbase = mt * 32, nbase = nt * 64;
    int ln = lane & 15, hi = lane >> 4;

    v8f acc[2][4] = {};
    const _Float16* a0 = A + (mbase + ln) * K + hi * 8;   // A lane: row, interleaved K
    const _Float16* a1 = a0 + 16 * K;

    for (int kb = 0; kb < K; kb += 32) {
        V16H av0, av1;
        av0.h[0] = *(const v8h*)(a0 + kb);      av0.h[1] = *(const v8h*)(a0 + kb + 16);
        av1.h[0] = *(const v8h*)(a1 + kb);      av1.h[1] = *(const v8h*)(a1 + kb + 16);
        #pragma unroll
        for (int t = 0; t < 4; ++t) {
            const _Float16* wp = W + (nbase + t * 16 + ln) * K + kb + hi * 8;
            V16H bv; bv.h[0] = *(const v8h*)wp; bv.h[1] = *(const v8h*)(wp + 16);
            acc[0][t] = WMMA_F16(av0.v, bv.v, acc[0][t]);
            acc[1][t] = WMMA_F16(av1.v, bv.v, acc[1][t]);
        }
    }
    #pragma unroll
    for (int a = 0; a < 2; ++a)
        #pragma unroll
        for (int t = 0; t < 4; ++t) {
            int col = nbase + t * 16 + ln;
            float bb = FUSE ? bias[col] : 0.0f;
            #pragma unroll
            for (int v = 0; v < 8; ++v) {
                int row = mbase + a * 16 + v + 8 * hi;
                float val = acc[a][t][v] + bb;
                if (FUSE) val += resid[row * NC + col];
                C[row * NC + col] = val;
            }
        }
}

// ---------------------------------------------------------------- per-(b,c) mean / unbiased std over n
__global__ void k_qstats(const float* __restrict__ q, float* __restrict__ mean,
                         float* __restrict__ stdv) {
    __shared__ float ssum[8][32], ssq[8][32];
    int b = blockIdx.x / 20, cg = blockIdx.x % 20;
    int tc = threadIdx.x & 31, tr = threadIdx.x >> 5;
    int c = cg * 32 + tc;
    const float* p = q + (size_t)b * NN * NC + c;
    float s = 0.f, s2 = 0.f;
    for (int r = tr; r < NN; r += 8) { float x = p[r * NC]; s += x; s2 += x * x; }
    ssum[tr][tc] = s; ssq[tr][tc] = s2;
    __syncthreads();
    if (tr == 0) {
        for (int i = 1; i < 8; ++i) { s += ssum[i][tc]; s2 += ssq[i][tc]; }
        float m = s / (float)NN;
        float var = (s2 - (float)NN * m * m) / (float)(NN - 1);   // ddof=1
        mean[b * NC + c] = m;
        stdv[b * NC + c] = sqrtf(var + 1e-5f);
    }
}

// ---------------------------------------------------------------- AdaIN apply to Q (+ fold ATTN_SCALE) -> f16
__global__ void k_qadain(const float* __restrict__ q, const float* __restrict__ mean,
                         const float* __restrict__ stdv, _Float16* __restrict__ qh) {
    int i = blockIdx.x * blockDim.x + threadIdx.x;
    if (i >= NB * NN * NC) return;
    int c = i % NC;
    int b = i / (NN * NC);
    int sb = (b < 2) ? 0 : 2;
    float m = mean[b * NC + c], sd = stdv[b * NC + c];
    float sm = mean[sb * NC + c], ss = stdv[sb * NC + c];
    qh[i] = (_Float16)((((q[i] - m) / sd) * ss + sm) * 0.125f);   // 64^-0.5
}

// ---------------------------------------------------------------- V transpose: [kvb,n,c] f32 -> [kvb,h,d,n] f16
__global__ void k_vtrans(const float* __restrict__ vf, _Float16* __restrict__ vt) {
    int i = blockIdx.x * blockDim.x + threadIdx.x;
    if (i >= 2 * NC * NN) return;
    int n   = i % NN;
    int dc  = (i / NN) % NC;        // h*64 + d
    int kvb = i / (NN * NC);
    vt[i] = (_Float16)vf[((size_t)kvb * NN + n) * NC + dc];
}

// ---------------------------------------------------------------- flash attention, transposed-S formulation
// one wave = 16 q columns; S^T = K·Q^T so softmax stats are (almost) lane-local,
// P^T feeds the O^T = V^T·P^T WMMA as B operand with only register shuffles.
__global__ void k_attn(const _Float16* __restrict__ qh, const _Float16* __restrict__ kh,
                       const _Float16* __restrict__ vt, _Float16* __restrict__ aoh) {
    int wv = threadIdx.x >> 5, lane = threadIdx.x & 31;
    int gw = blockIdx.x * 4 + wv;                        // 5120 wave-tiles exactly
    int b  = gw / (NH * 128);
    int h  = (gw / 128) % NH;
    int qt = gw % 128;
    int kvb = (b < 2) ? 0 : 1;
    int ln = lane & 15, hi = lane >> 4;

    // Q^T B-operand: lane = q column, K-dims interleaved; resident for whole loop
    const _Float16* qp = qh + (size_t)(b * NN + qt * 16 + ln) * NC + h * HD + hi * 8;
    V16H bQ0, bQ1;
    bQ0.h[0] = *(const v8h*)(qp);      bQ0.h[1] = *(const v8h*)(qp + 16);  // dims 0..31
    bQ1.h[0] = *(const v8h*)(qp + 32); bQ1.h[1] = *(const v8h*)(qp + 48);  // dims 32..63

    v8f o[4] = {};                       // O^T: 64 dims x 16 q (lane = q col)
    float rm = -__builtin_inff();        // per-lane running max (combined via lane^16)
    float rs = 0.f;                      // per-lane partial row sum

    const _Float16* kbase = kh + (size_t)kvb * NN * NC + h * HD;
    const _Float16* vbase = vt + ((size_t)(kvb * NH + h) * HD) * NN;

    for (int kb = 0; kb < NN; kb += 32) {
        // ---- S^T tiles: keys kb..kb+15 (s0), kb+16..kb+31 (s1); A = K rows (interleaved K-dims)
        v8f s0 = {}, s1 = {};
        {
            const _Float16* kp = kbase + (size_t)(kb + ln) * NC + hi * 8;
            V16H a0, a1;
            a0.h[0] = *(const v8h*)kp;        a0.h[1] = *(const v8h*)(kp + 16);
            a1.h[0] = *(const v8h*)(kp + 32); a1.h[1] = *(const v8h*)(kp + 48);
            s0 = WMMA_F16(a0.v, bQ0.v, s0);
            s0 = WMMA_F16(a1.v, bQ1.v, s0);
        }
        {
            const _Float16* kp = kbase + (size_t)(kb + 16 + ln) * NC + hi * 8;
            V16H a0, a1;
            a0.h[0] = *(const v8h*)kp;        a0.h[1] = *(const v8h*)(kp + 16);
            a1.h[0] = *(const v8h*)(kp + 32); a1.h[1] = *(const v8h*)(kp + 48);
            s1 = WMMA_F16(a0.v, bQ0.v, s1);
            s1 = WMMA_F16(a1.v, bQ1.v, s1);
        }
        // ---- online softmax: lane-local max over 16 keys + one lane^16 combine
        float mx = fmaxf(s0[0], s1[0]);
        #pragma unroll
        for (int v = 1; v < 8; ++v) mx = fmaxf(mx, fmaxf(s0[v], s1[v]));
        mx = fmaxf(mx, __shfl_xor(mx, 16, 32));
        float nm = fmaxf(rm, mx);
        float corr = __expf(rm - nm);
        rm = nm;
        float p0[8], p1[8], ls = 0.f;
        #pragma unroll
        for (int v = 0; v < 8; ++v) {
            p0[v] = __expf(s0[v] - nm);
            p1[v] = __expf(s1[v] - nm);
            ls += p0[v] + p1[v];
        }
        rs = rs * corr + ls;
        #pragma unroll
        for (int d = 0; d < 4; ++d)
            #pragma unroll
            for (int v = 0; v < 8; ++v) o[d][v] *= corr;     // per-lane scalar rescale

        // ---- build P^T B-operand (32 keys x 16 q): pack pairs + lane^16 exchange
        BP16 bP;
        unsigned pk0[4], pk1[4];
        #pragma unroll
        for (int j = 0; j < 4; ++j) {
            pk0[j] = pkf16(p0[2 * j], p0[2 * j + 1]);
            pk1[j] = pkf16(p1[2 * j], p1[2 * j + 1]);
        }
        #pragma unroll
        for (int j = 0; j < 4; ++j) {
            unsigned r0 = (unsigned)__shfl_xor((int)pk0[j], 16, 32);
            unsigned r1 = (unsigned)__shfl_xor((int)pk1[j], 16, 32);
            bP.u[j]     = hi ? r1     : pk0[j];   // keys 0-7  | 16-23
            bP.u[4 + j] = hi ? pk1[j] : r0;       // keys 8-15 | 24-31
        }
        // ---- O^T += V^T · P^T  (A = V^T tile, contiguous b128 loads from vt)
        #pragma unroll
        for (int d = 0; d < 4; ++d) {
            const _Float16* vp = vbase + (size_t)(d * 16 + ln) * NN + kb + hi * 16;
            V16H av; av.h[0] = *(const v8h*)vp; av.h[1] = *(const v8h*)(vp + 8);
            o[d] = WMMA_F16(av.v, bP.v, o[d]);
        }
    }
    // ---- finalize: combine lane-pair row sums, normalize, packed b128 stores
    float rst = rs + __shfl_xor(rs, 16, 32);
    float rinv = 1.0f / rst;
    _Float16* op = aoh + (size_t)(b * NN + qt * 16 + ln) * NC + h * HD + 8 * hi;
    #pragma unroll
    for (int d = 0; d < 4; ++d) {
        v8h ov;
        #pragma unroll
        for (int v = 0; v < 8; ++v) ov[v] = (_Float16)(o[d][v] * rinv);
        *(v8h*)(op + d * 16) = ov;     // dims d*16+8*hi .. +7, contiguous
    }
}

// ---------------------------------------------------------------- launcher
extern "C" void kernel_launch(void* const* d_in, const int* in_sizes, int n_in,
                              void* d_out, int out_size, void* d_ws, size_t ws_size,
                              hipStream_t stream) {
    const float* x  = (const float*)d_in[0];
    const float* wq = (const float*)d_in[1];
    const float* wk = (const float*)d_in[2];
    const float* wv = (const float*)d_in[3];
    const float* wo = (const float*)d_in[4];
    const float* bo = (const float*)d_in[5];

    char* ws = (char*)d_ws;
    const size_t SZ_X16 = (size_t)NB * NN * NC * 2;      // 10.5 MB
    const size_t SZ_W16 = (size_t)NC * NC * 2;           // 0.82 MB
    const size_t SZ_QF  = (size_t)NB * NN * NC * 4;      // 21 MB
    const size_t SZ_T2F = (size_t)2 * NN * NC * 4;       // 10.5 MB (2-batch f32 tmp)
    const size_t SZ_H2  = (size_t)2 * NN * NC * 2;       // 5.2 MB

    size_t off = 0;
    _Float16* xh  = (_Float16*)(ws + off); off += SZ_X16;
    _Float16* wqh = (_Float16*)(ws + off); off += SZ_W16;
    _Float16* wkh = (_Float16*)(ws + off); off += SZ_W16;
    _Float16* wvh = (_Float16*)(ws + off); off += SZ_W16;
    _Float16* woh = (_Float16*)(ws + off); off += SZ_W16;
    float*    qf  = (float*)   (ws + off); off += SZ_QF;
    float*    tmp = (float*)   (ws + off); off += SZ_T2F;  // k f32, then v f32
    _Float16* qhh = (_Float16*)(ws + off); off += SZ_X16;
    _Float16* khh = (_Float16*)(ws + off); off += SZ_H2;
    _Float16* vth = (_Float16*)(ws + off); off += SZ_H2;
    _Float16* aoh = (_Float16*)(ws + off); off += SZ_X16;
    float*    mn  = (float*)   (ws + off); off += (size_t)NB * NC * 4;
    float*    sd  = (float*)   (ws + off); off += (size_t)NB * NC * 4;

    const int nX  = NB * NN * NC;                // 5,242,880
    const int nW  = NC * NC;                     // 409,600
    const int nH2 = 2 * NN * NC;                 // 2,621,440

    // 1) converts to f16
    k_tof16<<<(nX + 255) / 256, 256, 0, stream>>>(x,  xh,  nX);
    k_tof16<<<(nW + 255) / 256, 256, 0, stream>>>(wq, wqh, nW);
    k_tof16<<<(nW + 255) / 256, 256, 0, stream>>>(wk, wkh, nW);
    k_tof16<<<(nW + 255) / 256, 256, 0, stream>>>(wv, wvh, nW);
    k_tof16<<<(nW + 255) / 256, 256, 0, stream>>>(wo, woh, nW);

    auto gemm_blocks = [](int M) { int tiles = (M / 32) * (NC / 64); return (tiles + 3) / 4; };

    // 2) Q projection (all batches)
    k_gemm<false><<<gemm_blocks(NB * NN), 128, 0, stream>>>(xh, wqh, qf, NB * NN, nullptr, nullptr);
    // 3) Q AdaIN stats + apply (+fold attn scale) -> qhh
    k_qstats<<<NB * 20, 256, 0, stream>>>(qf, mn, sd);
    k_qadain<<<(nX + 255) / 256, 256, 0, stream>>>(qf, mn, sd, qhh);
    // 4) K projection: effective K = raw k of batches 0 and 2 only (adain is identity there)
    k_gemm<false><<<gemm_blocks(NN), 128, 0, stream>>>(xh,               wkh, tmp,           NN, nullptr, nullptr);
    k_gemm<false><<<gemm_blocks(NN), 128, 0, stream>>>(xh + 2 * NN * NC, wkh, tmp + NN * NC, NN, nullptr, nullptr);
    k_tof16<<<(nH2 + 255) / 256, 256, 0, stream>>>(tmp, khh, nH2);
    // 5) V projection (batches 0, 2) + transpose to [kvb,h,d,n] f16
    k_gemm<false><<<gemm_blocks(NN), 128, 0, stream>>>(xh,               wvh, tmp,           NN, nullptr, nullptr);
    k_gemm<false><<<gemm_blocks(NN), 128, 0, stream>>>(xh + 2 * NN * NC, wvh, tmp + NN * NC, NN, nullptr, nullptr);
    k_vtrans<<<(nH2 + 255) / 256, 256, 0, stream>>>(tmp, vth);
    // 6) flash attention: 4*10*128 = 5120 wave-tiles, 4 waves/block
    k_attn<<<1280, 128, 0, stream>>>(qhh, khh, vth, aoh);
    // 7) output projection + bias + residual -> d_out (f32)
    k_gemm<true><<<gemm_blocks(NB * NN), 128, 0, stream>>>(aoh, woh, (float*)d_out, NB * NN, bo, x);

    (void)in_sizes; (void)n_in; (void)out_size; (void)ws_size;
}
